// LSTMLayer_88115549045324
// MI455X (gfx1250) — matrix operand order repaired and checked by reading, not compiled, not run on hardware
//
#include <hip/hip_runtime.h>
#include <hip/hip_bf16.h>

typedef __attribute__((ext_vector_type(16))) __bf16 v16bf;
typedef __attribute__((ext_vector_type(8)))  __bf16 v8bf;
typedef __attribute__((ext_vector_type(8)))  float  v8f;

namespace {
constexpr int T = 512, B = 32, I = 1024, H = 1024;
constexpr int G4 = 4 * H;               // 4096 gate columns
constexpr int NBLK = 128;               // persistent blocks
constexpr int HC = H / NBLK;            // 8 hidden units per block
constexpr int THREADS = 256;            // 8 waves: {x,h} x (mt,nt) in 2x2x2

// ---- workspace layout (bytes) ----
constexpr size_t OFF_XB   = 0;                        // x in bf16, (T,B,I)
constexpr size_t SZ_XB    = (size_t)T * B * I * 2;    // 32 MB
constexpr size_t OFF_WIH  = OFF_XB + SZ_XB;           // W_ih bf16, (4H, I) row-major
constexpr size_t SZ_W     = (size_t)G4 * I * 2;       // 8 MB
constexpr size_t OFF_WHH  = OFF_WIH + SZ_W;           // W_hh bf16, (4H, H)
constexpr size_t OFF_H0   = OFF_WHH + SZ_W;           // h ping buffer bf16 (B,H)
constexpr size_t SZ_HB    = (size_t)B * H * 2;        // 64 KB
constexpr size_t OFF_H1   = OFF_H0 + SZ_HB;           // h pong buffer
constexpr size_t OFF_BS   = OFF_H1 + SZ_HB;           // bias sum f32 (4H)
constexpr size_t SZ_BS    = (size_t)G4 * 4;
constexpr size_t OFF_SYNC = OFF_BS + SZ_BS;           // per-step arrival counters (T u32)

__device__ __forceinline__ float sigmoidf_fast(float x) {
    return 1.0f / (1.0f + __expf(-x));
}
} // namespace

// ---------------------------------------------------------------------------
// Kernel 1: bf16 conversions of x / W_ih / W_hh / h0, bias sum, and zeroing
// the per-step sync counters. Grid-stride over concatenated segments.
// ---------------------------------------------------------------------------
__global__ void lstm_prep_kernel(const float* __restrict__ x,
                                 const float* __restrict__ h0,
                                 const float* __restrict__ W_ih,
                                 const float* __restrict__ W_hh,
                                 const float* __restrict__ b_ih,
                                 const float* __restrict__ b_hh,
                                 __bf16* __restrict__ xb,
                                 __bf16* __restrict__ wih,
                                 __bf16* __restrict__ whh,
                                 __bf16* __restrict__ hbuf0,
                                 float* __restrict__ bsum,
                                 unsigned* __restrict__ syncc) {
    const long NX = (long)T * B * I;
    const long NW = (long)G4 * I;
    const long NH = (long)B * H;
    const long total = NX + 2 * NW + NH + G4 + T;
    long idx = (long)blockIdx.x * blockDim.x + threadIdx.x;
    long stride = (long)gridDim.x * blockDim.x;
    for (long i = idx; i < total; i += stride) {
        if (i < NX) {
            xb[i] = (__bf16)x[i];
        } else if (i < NX + NW) {
            long j = i - NX;
            wih[j] = (__bf16)W_ih[j];
        } else if (i < NX + 2 * NW) {
            long j = i - NX - NW;
            whh[j] = (__bf16)W_hh[j];
        } else if (i < NX + 2 * NW + NH) {
            long j = i - NX - 2 * NW;
            hbuf0[j] = (__bf16)h0[j];
        } else if (i < NX + 2 * NW + NH + G4) {
            long j = i - NX - 2 * NW - NH;
            bsum[j] = b_ih[j] + b_hh[j];
        } else {
            syncc[i - NX - 2 * NW - NH - G4] = 0u;
        }
    }
}

// ---------------------------------------------------------------------------
// Kernel 2: persistent LSTM recurrence. 128 blocks x 8 waves. Each block owns
// 8 hidden units (32 gate columns [i0..7 | f0..7 | g0..7 | o0..7]); both
// weight slices live in LDS for the whole sequence. The K-reduction is split
// across wave-halves: waves 0-3 compute the x-projection partial, waves 4-7
// the h-recurrence partial (independent WMMA streams, 2 waves per SIMD32);
// the epilogue sums the two partials. One device-wide barrier per timestep
// with a double-buffered bf16 h vector in global memory.
// ---------------------------------------------------------------------------
__global__ __launch_bounds__(THREADS, 1)
void lstm_persistent_kernel(const __bf16* __restrict__ xb,
                            const __bf16* __restrict__ wih_g,
                            const __bf16* __restrict__ whh_g,
                            __bf16* __restrict__ hbuf0,
                            __bf16* __restrict__ hbuf1,
                            const float* __restrict__ bsum,
                            const float* __restrict__ c0,
                            float* __restrict__ out,
                            unsigned* __restrict__ syncc) {
    // LDS: 64KB + 64KB + 2*4KB + 1KB + 128B  (~137 KB of the 320 KB WGP pool)
    __shared__ __align__(16) __bf16 sWih[32 * I];
    __shared__ __align__(16) __bf16 sWhh[32 * H];
    __shared__ __align__(16) float  sGateX[32 * 32];
    __shared__ __align__(16) float  sGateH[32 * 32];
    __shared__ float sC[B * HC];
    __shared__ float sBias[32];

    const int tid  = threadIdx.x;
    const int lane = tid & 31;
    const int ln   = lane & 15;        // N (or M) index within tile
    const int kh   = lane >> 4;        // K-half select
    const int w    = tid >> 5;         // wave 0..7
    const int mt   = w & 1;            // M tile (batch 0-15 / 16-31)
    const int nt   = (w >> 1) & 1;     // N tile (gate cols 0-15 / 16-31)
    const int part = w >> 2;           // 0 = x-projection, 1 = h-recurrence
    const int j0   = blockIdx.x * HC;

    // --- preload weight slices into LDS (local col c -> global gate row n) ---
    for (int idx = tid; idx < 32 * (I / 2); idx += THREADS) {
        const int c  = idx >> 9;           // 0..31
        const int k4 = idx & 511;          // uint index within the 1024-col row
        const int n  = (c >> 3) * H + j0 + (c & 7);
        ((unsigned*)sWih)[c * (I / 2) + k4] =
            ((const unsigned*)(wih_g + (size_t)n * I))[k4];
        ((unsigned*)sWhh)[c * (H / 2) + k4] =
            ((const unsigned*)(whh_g + (size_t)n * H))[k4];
    }
    if (tid < 32) {
        const int n = (tid >> 3) * H + j0 + (tid & 7);
        sBias[tid] = bsum[n];
    }
    for (int e = tid; e < B * HC; e += THREADS) {
        const int m = e >> 3, u = e & 7;
        sC[e] = c0[m * H + j0 + u];
    }
    __syncthreads();

    const int kbase8 = kh * 16;
    // bias folded into the x-partial accumulator init (per D column)
    const float bcol = sBias[nt * 16 + ln];
    const __bf16* __restrict__ sW   = part ? sWhh : sWih;
    float* __restrict__       sGate = part ? sGateH : sGateX;
    const __bf16* __restrict__ brow = &sW[(nt * 16 + ln) * 1024 + kh * 16];

    for (int t = 0; t < T; ++t) {
        const __bf16* __restrict__ hcur = (t & 1) ? hbuf1 : hbuf0;
        __bf16* __restrict__ hnext      = (t & 1) ? hbuf0 : hbuf1;

        // A-operand row pointer: x_t rows for part 0, h rows for part 1
        const __bf16* __restrict__ arow =
            part ? (hcur + (size_t)(mt * 16 + ln) * H)
                 : (xb + ((size_t)t * B + (mt * 16 + ln)) * I);

        v8f acc = {};
        if (!part) {
#pragma unroll
            for (int r = 0; r < 8; ++r) acc[r] = bcol;
        }

        // ---- partial = A[M,K=1024] @ W[K,N] (bf16 WMMA, f32 accumulate) ----
#pragma unroll 4
        for (int kk = 0; kk < 1024 / 32; ++kk) {
            const int k0 = kk * 32;
            v8bf lo = *(const v8bf*)(arow + k0 + kbase8);
            v8bf hi = *(const v8bf*)(arow + k0 + 8 + kbase8);
            v16bf a = __builtin_shufflevector(lo, hi, 0, 1, 2, 3, 4, 5, 6, 7,
                                              8, 9, 10, 11, 12, 13, 14, 15);
            v16bf b = *(const v16bf*)(brow + k0);
            acc = __builtin_amdgcn_wmma_f32_16x16x32_bf16(
                false, a, false, b, (short)0, acc, false, false);
        }

        // x-waves: next step's A rows are known now -> prefetch across barrier
        if (!part && t + 1 < T) {
            const __bf16* nxt = xb + ((size_t)(t + 1) * B + (mt * 16 + ln)) * I;
            __builtin_prefetch(nxt, 0, 1);
            __builtin_prefetch(nxt + 512, 0, 1);
        }

        // ---- spill tile partial to LDS (C/D layout: row = r + 8*kh) ----
#pragma unroll
        for (int r = 0; r < 8; ++r) {
            sGate[(mt * 16 + r + 8 * kh) * 32 + nt * 16 + ln] = acc[r];
        }
        __syncthreads();

        // ---- fused gate nonlinearities + cell/hidden update (1 elem/thread) --
        {
            const int e = tid;                 // 256 elements == 256 threads
            const int m = e >> 3, u = e & 7;
            const float* gx = &sGateX[m * 32];
            const float* gh = &sGateH[m * 32];
            float ig = sigmoidf_fast(gx[u]      + gh[u]);
            float fg = sigmoidf_fast(gx[8 + u]  + gh[8 + u]);
            float gg = tanhf(        gx[16 + u] + gh[16 + u]);
            float og = sigmoidf_fast(gx[24 + u] + gh[24 + u]);
            float cn = fg * sC[e] + ig * gg;
            float hn = og * tanhf(cn);
            sC[e] = cn;
            hnext[m * H + j0 + u] = (__bf16)hn;
            out[((size_t)t * B + m) * H + j0 + u] = hn;
            if (t == T - 1) {
                out[(size_t)T * B * H + m * H + j0 + u] = hn;                 // h_f
                out[(size_t)T * B * H + (size_t)B * H + m * H + j0 + u] = cn; // c_f
            }
        }

        // ---- device-wide barrier: make all blocks' h_next visible ----
        __threadfence();
        __syncthreads();
        if (tid == 0) {
            __hip_atomic_fetch_add(&syncc[t], 1u, __ATOMIC_RELEASE,
                                   __HIP_MEMORY_SCOPE_AGENT);
            while (__hip_atomic_load(&syncc[t], __ATOMIC_ACQUIRE,
                                     __HIP_MEMORY_SCOPE_AGENT) < (unsigned)NBLK) {
                __builtin_amdgcn_s_sleep(2);
            }
        }
        __syncthreads();
    }
}

extern "C" void kernel_launch(void* const* d_in, const int* in_sizes, int n_in,
                              void* d_out, int out_size, void* d_ws, size_t ws_size,
                              hipStream_t stream) {
    (void)in_sizes; (void)n_in; (void)out_size; (void)ws_size;
    const float* x    = (const float*)d_in[0];
    const float* h0   = (const float*)d_in[1];
    const float* c0   = (const float*)d_in[2];
    const float* W_ih = (const float*)d_in[3];
    const float* W_hh = (const float*)d_in[4];
    const float* b_ih = (const float*)d_in[5];
    const float* b_hh = (const float*)d_in[6];
    float* out = (float*)d_out;

    char* ws = (char*)d_ws;
    __bf16*   xb    = (__bf16*)(ws + OFF_XB);
    __bf16*   wih   = (__bf16*)(ws + OFF_WIH);
    __bf16*   whh   = (__bf16*)(ws + OFF_WHH);
    __bf16*   hbuf0 = (__bf16*)(ws + OFF_H0);
    __bf16*   hbuf1 = (__bf16*)(ws + OFF_H1);
    float*    bsum  = (float*)(ws + OFF_BS);
    unsigned* syncc = (unsigned*)(ws + OFF_SYNC);

    lstm_prep_kernel<<<2048, 256, 0, stream>>>(x, h0, W_ih, W_hh, b_ih, b_hh,
                                               xb, wih, whh, hbuf0, bsum, syncc);

    lstm_persistent_kernel<<<NBLK, THREADS, 0, stream>>>(
        xb, wih, whh, hbuf0, hbuf1, bsum, c0, out, syncc);
}
